// SequenceGenerator_55405078118876
// MI455X (gfx1250) — matrix operand order repaired and checked by reading, not compiled, not run on hardware
//
#include <hip/hip_runtime.h>

// ---------------------------------------------------------------------------
// CDNA5 (gfx1250, wave32) WMMA types: v_wmma_f32_16x16x32_bf16
// ---------------------------------------------------------------------------
typedef __attribute__((ext_vector_type(16))) __bf16 v16bf;
typedef __attribute__((ext_vector_type(8)))  __bf16 v8bf;
typedef __attribute__((ext_vector_type(2)))  __bf16 v2bf;
typedef __attribute__((ext_vector_type(8)))  float  v8f;

#define LDA_PAD 40   // bf16 elems per LDS row (32 data + 8 pad, rows stay 16B aligned)
#define LDB_PAD 40

// ---------------------------------------------------------------------------
// bf16-WMMA GEMM:  C[M,N] = A[M,K] @ B[K,N] (+ bias[N]);  M is always 128.
//  - block tile 128(M) x 64(N), K chunk 32, double-buffered LDS pipeline
//  - 8 waves: wave>>2 -> M half (4 wmma M-subtiles), wave&3 -> 16-col N subtile
//  - optional split-K over gridDim.y (atomicAdd into pre-zeroed C)
//  - fast unguarded staging path for interior chunks/blocks
// ---------------------------------------------------------------------------
__global__ __launch_bounds__(256) void sg_gemm_bf16_wmma(
    const float* __restrict__ A, const float* __restrict__ Bw,
    const float* __restrict__ bias, float* __restrict__ C,
    int N, int K)
{
    __shared__ __align__(16) __bf16 ldsA[2][128 * LDA_PAD]; // [m][k] 128x32 used
    __shared__ __align__(16) __bf16 ldsB[2][64  * LDB_PAD]; // [n][k] 64x32 used (B^T)

    const int tid   = threadIdx.x;
    const int lane  = tid & 31;
    const int wave  = tid >> 5;
    const int wn    = wave & 3;
    const int wmh   = wave >> 2;
    const int nBase = blockIdx.x * 64;
    const bool nFull = (nBase + 64 <= N);

    v8f acc[4];
#pragma unroll
    for (int t = 0; t < 4; ++t) acc[t] = (v8f)0.f;

    const int nChunks  = (K + 31) >> 5;
    const int perSplit = (nChunks + gridDim.y - 1) / gridDim.y;
    const int c0 = blockIdx.y * perSplit;
    const int c1 = (c0 + perSplit < nChunks) ? (c0 + perSplit) : nChunks;

    // thread's fixed staging coordinates
    const int aRow = tid >> 1;            // pairs: 2 threads per row? (see loops below)

    auto stage = [&](int ci, int buf) {
        const int kc = ci << 5;
        __bf16* sA = &ldsA[buf][0];
        __bf16* sB = &ldsB[buf][0];
        if (kc + 32 <= K) {
            // ---------------- fast path: no K guards ----------------
#pragma unroll
            for (int i = 0; i < 8; ++i) {
                int p = tid + 256 * i;          // 2048 K-pairs of A (128x32)
                int row = p >> 4;
                int cp  = (p & 15) * 2;
                const float* ap = A + (size_t)row * K + (kc + cp);
                float f0 = ap[0];
                float f1 = ap[1];
                v2bf pr; pr[0] = (__bf16)f0; pr[1] = (__bf16)f1;
                *(v2bf*)&sA[row * LDA_PAD + cp] = pr;
            }
            if (nFull) {
#pragma unroll
                for (int i = 0; i < 8; ++i) {
                    int p = tid + 256 * i;      // 2048 K-pairs of B (32x64)
                    int col = p & 63;
                    int kp  = (p >> 6) * 2;
                    const float* bp = Bw + (size_t)(kc + kp) * N + nBase + col;
                    float f0 = __builtin_nontemporal_load(bp);
                    float f1 = __builtin_nontemporal_load(bp + N);
                    v2bf pr; pr[0] = (__bf16)f0; pr[1] = (__bf16)f1;
                    *(v2bf*)&sB[col * LDB_PAD + kp] = pr;
                }
            } else {
#pragma unroll
                for (int i = 0; i < 8; ++i) {
                    int p = tid + 256 * i;
                    int col = p & 63;
                    int kp  = (p >> 6) * 2;
                    int gn  = nBase + col;
                    float f0 = 0.f, f1 = 0.f;
                    if (gn < N) {
                        const float* bp = Bw + (size_t)(kc + kp) * N + gn;
                        f0 = __builtin_nontemporal_load(bp);
                        f1 = __builtin_nontemporal_load(bp + N);
                    }
                    v2bf pr; pr[0] = (__bf16)f0; pr[1] = (__bf16)f1;
                    *(v2bf*)&sB[col * LDB_PAD + kp] = pr;
                }
            }
        } else {
            // ---------------- tail chunk: full guards ----------------
#pragma unroll
            for (int i = 0; i < 8; ++i) {
                int p = tid + 256 * i;
                int row = p >> 4;
                int cp  = (p & 15) * 2;
                int gk  = kc + cp;
                const float* ap = A + (size_t)row * K + gk;
                float f0 = (gk     < K) ? ap[0] : 0.f;
                float f1 = (gk + 1 < K) ? ap[1] : 0.f;
                v2bf pr; pr[0] = (__bf16)f0; pr[1] = (__bf16)f1;
                *(v2bf*)&sA[row * LDA_PAD + cp] = pr;
            }
#pragma unroll
            for (int i = 0; i < 8; ++i) {
                int p = tid + 256 * i;
                int col = p & 63;
                int kp  = (p >> 6) * 2;
                int gn  = nBase + col;
                int gk  = kc + kp;
                float f0 = 0.f, f1 = 0.f;
                if (gn < N) {
                    const float* bp = Bw + (size_t)gk * N + gn;
                    if (gk     < K) f0 = bp[0];
                    if (gk + 1 < K) f1 = bp[N];
                }
                v2bf pr; pr[0] = (__bf16)f0; pr[1] = (__bf16)f1;
                *(v2bf*)&sB[col * LDB_PAD + kp] = pr;
            }
        }
    };

    (void)aRow;
    int buf = 0;
    if (c0 < c1) stage(c0, 0);
    __syncthreads();

    for (int ci = c0; ci < c1; ++ci) {
        // prefetch next chunk into the other buffer (overlaps with WMMA below)
        if (ci + 1 < c1) stage(ci + 1, buf ^ 1);

        const __bf16* sA = &ldsA[buf][0];
        const __bf16* sB = &ldsB[buf][0];

        // B fragment (32x16): elems 0..7 -> K=kbB..kbB+7, 8..15 -> K=kbB+16..+23, N = lane&15
        const int bn  = wn * 16 + (lane & 15);
        const int kbB = (lane < 16) ? 0 : 8;
        v8bf b0 = *(const v8bf*)&sB[bn * LDB_PAD + kbB];
        v8bf b1 = *(const v8bf*)&sB[bn * LDB_PAD + kbB + 16];
        v16bf bfrag = __builtin_shufflevector(b0, b1,
            0, 1, 2, 3, 4, 5, 6, 7, 8, 9, 10, 11, 12, 13, 14, 15);

        // A fragments (16x32): elems 0..7 -> K=kbA..kbA+7, 8..15 -> K=kbA+16..+23
        const int am  = wmh * 64 + (lane & 15);
        const int kbA = (lane < 16) ? 0 : 8;
#pragma unroll
        for (int t = 0; t < 4; ++t) {
            int m = am + t * 16;
            v8bf a0 = *(const v8bf*)&sA[m * LDA_PAD + kbA];
            v8bf a1 = *(const v8bf*)&sA[m * LDA_PAD + kbA + 16];
            v16bf afrag = __builtin_shufflevector(a0, a1,
                0, 1, 2, 3, 4, 5, 6, 7, 8, 9, 10, 11, 12, 13, 14, 15);
            acc[t] = __builtin_amdgcn_wmma_f32_16x16x32_bf16(
                false, afrag, false, bfrag, (short)0, acc[t], false, false);
        }
        __syncthreads();
        buf ^= 1;
    }

    // store per 16x16 f32 C/D layout: VGPR i -> M=i (lanes 0-15) / i+8 (lanes 16-31)
    const int cn = nBase + wn * 16 + (lane & 15);
    if (cn < N) {
        const bool split = (gridDim.y > 1);
        const float bv = (bias && (!split || blockIdx.y == 0)) ? bias[cn] : 0.f;
#pragma unroll
        for (int t = 0; t < 4; ++t) {
            const int mBase = wmh * 64 + t * 16;
#pragma unroll
            for (int i = 0; i < 8; ++i) {
                int m = mBase + ((lane < 16) ? i : i + 8);
                float v = acc[t][i] + bv;
                float* cp = &C[(size_t)m * N + cn];
                if (split) atomicAdd(cp, v);
                else       *cp = v;
            }
        }
    }
}

// ---------------------------------------------------------------------------
// GRU gates: h = z*hprev + (1-z)*tanh(xh + r*hh)
// ---------------------------------------------------------------------------
__global__ __launch_bounds__(256) void sg_gru_gate(
    const float* __restrict__ XM, const float* __restrict__ HM,
    const float* __restrict__ Hprev, float* __restrict__ Hout)
{
    int idx = blockIdx.x * 256 + threadIdx.x;   // 0 .. 128*1024-1
    int b = idx >> 10, j = idx & 1023;
    const float* xm = XM + (size_t)b * 3072;
    const float* hm = HM + (size_t)b * 3072;
    float xz = xm[j], xr = xm[1024 + j], xh = xm[2048 + j];
    float hz = hm[j], hr = hm[1024 + j], hh = hm[2048 + j];
    float z = 1.f / (1.f + __expf(-(xz + hz)));
    float r = 1.f / (1.f + __expf(-(xr + hr)));
    float c = tanhf(xh + r * hh);
    Hout[idx] = z * Hprev[idx] + (1.f - z) * c;
}

// ---------------------------------------------------------------------------
// Single-pass online softmax per row (one workgroup per row), then normalize.
// ---------------------------------------------------------------------------
__global__ __launch_bounds__(256) void sg_softmax_rows(float* __restrict__ P, int N)
{
    __shared__ float rm[256], rs[256];
    float* p = P + (size_t)blockIdx.x * N;
    int tid = threadIdx.x;

    float m = -3.0e38f, s = 0.f;
    for (int j = tid; j < N; j += 256) {
        float x  = p[j];
        float nm = fmaxf(m, x);
        s = s * __expf(m - nm) + __expf(x - nm);
        m = nm;
    }
    rm[tid] = m; rs[tid] = s; __syncthreads();
    for (int st = 128; st > 0; st >>= 1) {
        if (tid < st) {
            float m2 = rm[tid + st], s2 = rs[tid + st];
            float nm = fmaxf(rm[tid], m2);
            rs[tid] = rs[tid] * __expf(rm[tid] - nm) + s2 * __expf(m2 - nm);
            rm[tid] = nm;
        }
        __syncthreads();
    }
    float gm = rm[0], inv = 1.f / rs[0];
    for (int j = tid; j < N; j += 256) p[j] = __expf(p[j] - gm) * inv;
}

__global__ __launch_bounds__(256) void sg_zero(float* __restrict__ p, int n)
{
    int i = blockIdx.x * 256 + threadIdx.x;
    if (i < n) p[i] = 0.f;
}

// ---------------------------------------------------------------------------
extern "C" void kernel_launch(void* const* d_in, const int* in_sizes, int n_in,
                              void* d_out, int out_size, void* d_ws, size_t ws_size,
                              hipStream_t stream) {
    (void)in_sizes; (void)n_in; (void)out_size; (void)ws_size;
    const int B = 128, V = 50257, E = 512, U = 1024, U3 = 3072;

    const float* inputs = (const float*)d_in[0];
    const float* h0     = (const float*)d_in[1];
    const float* embedW = (const float*)d_in[2];
    const float* k0     = (const float*)d_in[3];
    const float* rk0    = (const float*)d_in[4];
    const float* b0     = (const float*)d_in[5];
    const float* k1     = (const float*)d_in[6];
    const float* rk1    = (const float*)d_in[7];
    const float* b1     = (const float*)d_in[8];
    const float* denseW = (const float*)d_in[9];
    const float* denseB = (const float*)d_in[10];

    float* ws = (float*)d_ws;
    float* S  = ws;                       // [B,E]
    float* XM = S  + (size_t)B * E;       // [B,3U]
    float* HM = XM + (size_t)B * U3;      // [B,3U]
    float* H1 = HM + (size_t)B * U3;      // [B,U]
    float* P  = (float*)d_out;            // [B,V] probs
    float* H2 = P + (size_t)B * V;        // [B,U] second output

    dim3 blk(256);

    // 1) S = inputs @ embed_W   (split-K x32 -> 256 blocks; atomicAdd into zeroed S)
    sg_zero<<<dim3((B * E + 255) / 256), blk, 0, stream>>>(S, B * E);
    sg_gemm_bf16_wmma<<<dim3(E / 64, 32), blk, 0, stream>>>(inputs, embedW, nullptr, S, E, V);

    // 2) XM = S @ k0 + b0[0] ; HM = h0 @ rk0 + b0[1]   (split-K x8 -> 384 blocks each)
    sg_zero<<<dim3((2 * B * U3 + 255) / 256), blk, 0, stream>>>(XM, 2 * B * U3);
    sg_gemm_bf16_wmma<<<dim3(U3 / 64, 8), blk, 0, stream>>>(S,  k0,  b0,      XM, U3, E);
    sg_gemm_bf16_wmma<<<dim3(U3 / 64, 8), blk, 0, stream>>>(h0, rk0, b0 + U3, HM, U3, U);

    // 3) h1 = gate(XM, HM, h0)
    sg_gru_gate<<<dim3(B * U / 256), blk, 0, stream>>>(XM, HM, h0, H1);

    // 4) XM = h1 @ k1 + b1[0] ; HM = h1 @ rk1 + b1[1]
    sg_zero<<<dim3((2 * B * U3 + 255) / 256), blk, 0, stream>>>(XM, 2 * B * U3);
    sg_gemm_bf16_wmma<<<dim3(U3 / 64, 8), blk, 0, stream>>>(H1, k1,  b1,      XM, U3, U);
    sg_gemm_bf16_wmma<<<dim3(U3 / 64, 8), blk, 0, stream>>>(H1, rk1, b1 + U3, HM, U3, U);

    // 5) h2 = gate(XM, HM, h1) -> d_out tail
    sg_gru_gate<<<dim3(B * U / 256), blk, 0, stream>>>(XM, HM, H1, H2);

    // 6) logits = h2 @ dense_W + dense_b   (786 blocks, weights streamed once, NT)
    sg_gemm_bf16_wmma<<<dim3((V + 63) / 64, 1), blk, 0, stream>>>(H2, denseW, denseB, P, V, U);

    // 7) probs = softmax(logits), single-pass stats + normalize
    sg_softmax_rows<<<dim3(B), blk, 0, stream>>>(P, V);
}
